// MSDNET_58987080843719
// MI455X (gfx1250) — matrix operand order repaired and checked by reading, not compiled, not run on hardware
//
#include <hip/hip_runtime.h>
#include <hip/hip_bf16.h>
#include <stdint.h>

typedef _Float16 v16h __attribute__((ext_vector_type(16)));
typedef _Float16 v8h  __attribute__((ext_vector_type(8)));
typedef float    v8f  __attribute__((ext_vector_type(8)));

#define DEPTH    100
#define HW       384
#define PAD      16
#define WP       416               // 384 + 2*PAD
#define PLANE    (416*416)
#define NCH_TOT  (DEPTH+1)
#define TILES_1D (HW/16)           // 24
#define NTILES   (2*TILES_1D*TILES_1D)
#define NW       4                 // waves per block (channel-split)

// ---------------- init: zero feats (incl. halos), fill channel 0 with x ----
__global__ void msd_zero_feats(_Float16* f, long n8) {
    long i = (long)blockIdx.x * blockDim.x + threadIdx.x;
    if (i < n8) {
        v8h z = {};
        ((v8h*)f)[i] = z;
    }
}

__global__ void msd_fill_x(const float* __restrict__ x, _Float16* feats) {
    int i = blockIdx.x * blockDim.x + threadIdx.x;
    if (i >= 2*HW*HW) return;
    int b = i / (HW*HW);
    int p = i % (HW*HW);
    int h = p / HW, w = p % HW;
    feats[(size_t)b*NCH_TOT*PLANE + (size_t)(h+PAD)*WP + (w+PAD)] = (_Float16)x[i];
}

// ---------------- pack banded-Toeplitz B matrices in WMMA operand order ----
// Chunk = 32(K) x 16(N) f16 = 512 halves, stored as [lane 0..31][16 halves],
// lane = n + 16*g, element h -> K = ci*32 + g*16 + h.
// B[K][n] = w[c][kh][kw] iff (j0+K) - n == (kw-1)*dil, else 0.
__global__ void msd_pack_b(const float* __restrict__ w, _Float16* dst,
                           int dil, int nch, int j0, int nelem) {
    int t = blockIdx.x * blockDim.x + threadIdx.x;
    if (t >= nelem) return;
    int chunk  = t >> 9;
    int within = t & 511;
    int lane = within >> 4;
    int h    = within & 15;
    int ci = chunk % nch;
    int q  = chunk / nch;
    int kh = q % 3;
    int c  = q / 3;
    int n = lane & 15;
    int g = lane >> 4;
    int K = ci*32 + g*16 + h;
    int delta = (j0 + K) - n;
    const float* wr = w + ((size_t)c*3 + kh)*3;   // (c, kh, kw)
    float val = 0.0f;
    if      (delta == -dil) val = wr[0];
    else if (delta ==  0  ) val = wr[1];
    else if (delta ==  dil) val = wr[2];
    dst[(size_t)chunk*512 + lane*16 + h] = (_Float16)val;
}

// ---------------- one dense layer: implicit banded GEMM via WMMA -----------
// 4 waves take contiguous channel ranges; each channel's banded-B block
// (3*NCHUNK KB, contiguous) is double-buffered through LDS via async
// global->LDS copies; partial f32 accumulators reduced through LDS.
template<int NCHUNK>
__global__ __launch_bounds__(32*NW) void msd_conv_layer(
    _Float16* feats, const _Float16* __restrict__ packB,
    int nchan, int dil, int j0, int layer,
    const float* __restrict__ conv_bs, const float* __restrict__ bn_scale,
    const float* __restrict__ bn_bias, const float* __restrict__ bn_mean,
    const float* __restrict__ bn_var)
{
    constexpr int NCK = 3*NCHUNK;        // chunks per channel
    constexpr int CB  = NCK*1024;        // bytes per channel B block
    __shared__ __align__(32) unsigned char bstage[NW][2][CB];
    __shared__ float red[NW][256];

    int tile = blockIdx.x;
    int b  = tile / (TILES_1D*TILES_1D);
    int t2 = tile % (TILES_1D*TILES_1D);
    int rowBase = (t2 / TILES_1D) * 16;
    int colBase = (t2 % TILES_1D) * 16;

    int tid  = threadIdx.x;
    int w    = __builtin_amdgcn_readfirstlane(tid >> 5);  // scalar wave id
    int lane = tid & 31;
    int lo = lane & 15;      // A: row m ; D: column n
    int g  = lane >> 4;      // A/B: K half-select ; D: m high bit

    const _Float16* fb = feats + (size_t)b * NCH_TOT * PLANE;
    const char* packBb = (const char*)packB;

    int c0 = __builtin_amdgcn_readfirstlane((nchan * w)       / NW);
    int c1 = __builtin_amdgcn_readfirstlane((nchan * (w + 1)) / NW);

    unsigned ldsBase0 = (unsigned)(size_t)(&bstage[w][0][0]) + (unsigned)lane*16u;

    // async-stage one channel's B block (NCK KB) into an LDS buffer
    auto issueB = [&](int c, unsigned ldsb) {
        uint64_t ga = (uint64_t)(packBb + (size_t)c*CB + lane*16);
        if constexpr (NCHUNK == 1) {
            asm volatile(
                "global_load_async_to_lds_b128 %0, %1, off\n\t"
                "global_load_async_to_lds_b128 %0, %1, off offset:512\n\t"
                "global_load_async_to_lds_b128 %0, %1, off offset:1024\n\t"
                "global_load_async_to_lds_b128 %0, %1, off offset:1536\n\t"
                "global_load_async_to_lds_b128 %0, %1, off offset:2048\n\t"
                "global_load_async_to_lds_b128 %0, %1, off offset:2560"
                :: "v"(ldsb), "v"(ga));
        } else {
            asm volatile(
                "global_load_async_to_lds_b128 %0, %1, off\n\t"
                "global_load_async_to_lds_b128 %0, %1, off offset:512\n\t"
                "global_load_async_to_lds_b128 %0, %1, off offset:1024\n\t"
                "global_load_async_to_lds_b128 %0, %1, off offset:1536\n\t"
                "global_load_async_to_lds_b128 %0, %1, off offset:2048\n\t"
                "global_load_async_to_lds_b128 %0, %1, off offset:2560\n\t"
                "global_load_async_to_lds_b128 %0, %1, off offset:3072\n\t"
                "global_load_async_to_lds_b128 %0, %1, off offset:3584\n\t"
                "global_load_async_to_lds_b128 %0, %1, off offset:4096\n\t"
                "global_load_async_to_lds_b128 %0, %1, off offset:4608\n\t"
                "global_load_async_to_lds_b128 %0, %1, off offset:5120\n\t"
                "global_load_async_to_lds_b128 %0, %1, off offset:5632"
                :: "v"(ldsb), "v"(ga));
        }
    };

    v8f acc = {};
    int parity = 0;

    if (c0 < c1) issueB(c0, ldsBase0);   // prologue into buffer 0

    for (int c = c0; c < c1; ++c) {
        // ---- A operands for all chunks of this channel (issued first so the
        //      backend can clause them ahead of the asm fences) -------------
        v8h a0[NCK], a1[NCK];
        #pragma unroll
        for (int kh = 0; kh < 3; ++kh) {
            int rowp = rowBase + lo + (kh - 1)*dil + PAD;
            const _Float16* rp = fb + (size_t)c*PLANE + (size_t)rowp*WP + (colBase + PAD);
            #pragma unroll
            for (int ci = 0; ci < NCHUNK; ++ci) {
                int col0 = j0 + ci*32 + g*16;
                a0[kh*NCHUNK + ci] = *(const v8h*)(rp + col0);
                a1[kh*NCHUNK + ci] = *(const v8h*)(rp + col0 + 8);
            }
        }

        // ---- stage next channel's B; wait for current buffer --------------
        unsigned off = parity ? (unsigned)CB : 0u;   // byte offset of current buf
        if (c + 1 < c1) {
            issueB(c + 1, ldsBase0 + (parity ? 0u : (unsigned)CB));
            if constexpr (NCHUNK == 1)
                asm volatile("s_wait_asynccnt 0x6" : "+v"(off));
            else
                asm volatile("s_wait_asynccnt 0xc" : "+v"(off));
        } else {
            asm volatile("s_wait_asynccnt 0x0" : "+v"(off));
        }

        // ---- consume: B from LDS, WMMA chain ------------------------------
        const unsigned char* bufp = &bstage[w][0][0] + off;
        #pragma unroll
        for (int k = 0; k < NCK; ++k) {
            const v16h bm = *(const v16h*)(bufp + k*1024 + lane*32); // ds_load_b128 x2
            union { v16h v; v8h h2[2]; } au;
            au.h2[0] = a0[k];
            au.h2[1] = a1[k];
            acc = __builtin_amdgcn_wmma_f32_16x16x32_f16(
                      false, au.v, false, bm, (short)0, acc, false, false);
        }
        parity ^= 1;
    }

    // cross-wave reduction of partial accumulators through LDS
    #pragma unroll
    for (int r8 = 0; r8 < 8; ++r8)
        red[w][lane*8 + r8] = acc[r8];
    __syncthreads();

    if (w == 0) {
        float s = bn_scale[layer] * rsqrtf(bn_var[layer] + 1e-5f);
        float tt = (conv_bs[layer] - bn_mean[layer]) * s + bn_bias[layer];
        _Float16* op = feats + ((size_t)b*NCH_TOT + nchan) * PLANE;
        #pragma unroll
        for (int r8 = 0; r8 < 8; ++r8) {
            // D layout: lane L, vgpr r -> m = r + 8*(L>=16), n = L&15
            float v = red[0][lane*8+r8] + red[1][lane*8+r8]
                    + red[2][lane*8+r8] + red[3][lane*8+r8];
            v = v*s + tt;
            v = v > 0.0f ? v : 0.0f;
            op[(size_t)(rowBase + r8 + 8*g + PAD)*WP + (colBase + lo + PAD)] = (_Float16)v;
        }
    }
}

// ---------------- final 1x1 conv over 101 channels + sigmoid ---------------
__global__ void msd_final_head(const _Float16* __restrict__ feats,
                               const float* __restrict__ last_w,
                               const float* __restrict__ last_b,
                               float* __restrict__ out) {
    int i = blockIdx.x * blockDim.x + threadIdx.x;
    if (i >= 2*HW*HW) return;
    int b = i / (HW*HW);
    int p = i % (HW*HW);
    int h = p / HW, w = p % HW;
    const _Float16* fp = feats + (size_t)b*NCH_TOT*PLANE + (size_t)(h+PAD)*WP + (w+PAD);
    float a = last_b[0];
    #pragma unroll 4
    for (int c = 0; c < NCH_TOT; ++c)
        a += (float)fp[(size_t)c*PLANE] * last_w[c];
    out[i] = 1.0f / (1.0f + __expf(-a));
}

// ---------------------------------------------------------------------------
extern "C" void kernel_launch(void* const* d_in, const int* in_sizes, int n_in,
                              void* d_out, int out_size, void* d_ws, size_t ws_size,
                              hipStream_t stream) {
    (void)in_sizes; (void)out_size; (void)ws_size;

    const float* x = (const float*)d_in[0];
    // last 7 inputs are conv_bs, bn_scale, bn_bias, bn_mean, bn_var, last_w, last_b
    int tail = n_in - 7;
    const float* conv_bs  = (const float*)d_in[tail + 0];
    const float* bn_scale = (const float*)d_in[tail + 1];
    const float* bn_bias  = (const float*)d_in[tail + 2];
    const float* bn_mean  = (const float*)d_in[tail + 3];
    const float* bn_var   = (const float*)d_in[tail + 4];
    const float* last_w   = (const float*)d_in[tail + 5];
    const float* last_b   = (const float*)d_in[tail + 6];

    _Float16* feats = (_Float16*)d_ws;
    size_t featsBytes = (size_t)2 * NCH_TOT * PLANE * sizeof(_Float16);
    size_t bByteOff   = (featsBytes + 255) & ~(size_t)255;
    _Float16* packB   = (_Float16*)((char*)d_ws + bByteOff);

    long n8 = (long)2 * NCH_TOT * PLANE / 8;
    msd_zero_feats<<<(unsigned)((n8 + 255) / 256), 256, 0, stream>>>(feats, n8);
    int npix = 2 * HW * HW;
    msd_fill_x<<<(npix + 255) / 256, 256, 0, stream>>>(x, feats);

    // Pack all layers' banded B matrices (independent of feats).
    int concat = (n_in < 1 + DEPTH + 7);   // fallback: conv_ws flattened into d_in[1]
    size_t layerOff[DEPTH];
    size_t boffHalves = 0;
    size_t woff = 0;
    for (int i = 0; i < DEPTH; ++i) {
        int dil = 1 + i % 10;
        int nch = (dil <= 8) ? 1 : 2;
        int j0  = (nch == 1) ? -8 : -16;
        int chunks = (i + 1) * 3 * nch;
        int nelem  = chunks * 512;
        layerOff[i] = boffHalves;
        const float* wp = concat ? ((const float*)d_in[1] + woff)
                                 : (const float*)d_in[1 + i];
        msd_pack_b<<<(nelem + 255) / 256, 256, 0, stream>>>(
            wp, packB + boffHalves, dil, nch, j0, nelem);
        boffHalves += (size_t)nelem;
        woff += (size_t)9 * (i + 1);
    }

    // Sequential dense layers (grid-wide dependency -> one launch per layer).
    for (int i = 0; i < DEPTH; ++i) {
        int dil = 1 + i % 10;
        int nch = (dil <= 8) ? 1 : 2;
        int j0  = (nch == 1) ? -8 : -16;
        if (nch == 1)
            msd_conv_layer<1><<<NTILES, 32*NW, 0, stream>>>(
                feats, packB + layerOff[i], i + 1, dil, j0, i,
                conv_bs, bn_scale, bn_bias, bn_mean, bn_var);
        else
            msd_conv_layer<2><<<NTILES, 32*NW, 0, stream>>>(
                feats, packB + layerOff[i], i + 1, dil, j0, i,
                conv_bs, bn_scale, bn_bias, bn_mean, bn_var);
    }

    msd_final_head<<<(npix + 255) / 256, 256, 0, stream>>>(
        feats, last_w, last_b, (float*)d_out);
}
